// DGCNNMultiModalCondSSMT_83717502533974
// MI455X (gfx1250) — matrix operand order repaired and checked by reading, not compiled, not run on hardware
//
#include <hip/hip_runtime.h>
#include <hip/hip_bf16.h>

// ---------------------------------------------------------------------------
// DGCNN + cross-attn + Mamba pipeline for MI455X (gfx1250), wave32 + WMMA.
// GEMMs use v_wmma_f32_16x16x32_f16; weight panels <=48KB are staged into LDS
// with the Tensor Data Mover (tensor_load_to_lds + s_wait_tensorcnt).
// ---------------------------------------------------------------------------

typedef __attribute__((ext_vector_type(16))) _Float16 v16h;
typedef __attribute__((ext_vector_type(8)))  _Float16 v8h;
typedef __attribute__((ext_vector_type(8)))  float    v8f;
typedef __attribute__((ext_vector_type(4)))  unsigned int u32x4;
typedef __attribute__((ext_vector_type(4)))  int      i32x4;
typedef __attribute__((ext_vector_type(8)))  int      i32x8;

static constexpr int kB    = 4;
static constexpr int kBT   = 128;     // B*T frames
static constexpr int kN    = 384;     // points per frame
static constexpr int kK    = 20;      // KNN
static constexpr int kMEdge = kBT * kN * kK;  // 983040 rows (multiple of 128)
static constexpr int kNP    = kBT * kN;       // 49152 points
static constexpr int kWPanel = 24576;         // LDS weight panel (halves) = 48KB
static constexpr float kBnScale = 0.99999500003749969f; // (1+1e-5)^-0.5
static constexpr float kEps = 1e-5f;

// ---------------------------------------------------------------------------
// KNN: one wave per query point; LDS distance rows + wave32 shfl_xor reduce.
// ---------------------------------------------------------------------------
__global__ void knn_kernel(const float* __restrict__ x, int stride, int offs,
                           int C, int* __restrict__ idxOut)
{
    __shared__ float pts[kN * 32];
    __shared__ float dist[8 * kN];
    const int bt  = blockIdx.y;
    const int tid = threadIdx.x;
    for (int t = tid; t < kN * C; t += 256) {
        int n = t / C, c = t - n * C;
        pts[n * C + c] = x[((size_t)(bt * kN + n)) * stride + offs + c];
    }
    __syncthreads();
    const int wave = tid >> 5, lane = tid & 31;
    const int i = blockIdx.x * 8 + wave;
    for (int j = lane; j < kN; j += 32) {
        float d2 = 0.f;
        for (int c = 0; c < C; ++c) {
            float d = pts[i * C + c] - pts[j * C + c];
            d2 += d * d;
        }
        if (j == i) d2 += 1e9f;
        dist[wave * kN + j] = d2;
    }
    __syncthreads();
    for (int k = 0; k < kK; ++k) {
        float bv = 1e30f; int bi = 0;
        for (int j = lane; j < kN; j += 32) {
            float v = dist[wave * kN + j];
            if (v < bv) { bv = v; bi = j; }
        }
        for (int m = 16; m >= 1; m >>= 1) {
            float ov = __shfl_xor(bv, m, 32);
            int   oi = __shfl_xor(bi, m, 32);
            if (ov < bv || (ov == bv && oi < bi)) { bv = ov; bi = oi; }
        }
        if (lane == 0) {
            idxOut[((size_t)(bt * kN) + i) * kK + k] = bi;
            dist[wave * kN + bi] = 1e30f;
        }
        __syncthreads();
    }
}

// ---------------------------------------------------------------------------
// Gathered edge features in f16, zero-padded K. mode 0: [xi, xj-xi]; 1: [ai, aj]
// ---------------------------------------------------------------------------
__global__ void build_feat_kernel(const float* __restrict__ x, int stride,
                                  int offs, int C, const int* __restrict__ idx,
                                  _Float16* __restrict__ out, int Kpad, int mode)
{
    int row = blockIdx.x * 256 + threadIdx.x;
    if (row >= kMEdge) return;
    int p  = row / kK;
    int j  = idx[row];
    int bt = p / kN;
    const float* xi = x + (size_t)p * stride + offs;
    const float* xj = x + ((size_t)(bt * kN + j)) * stride + offs;
    _Float16* o = out + (size_t)row * Kpad;
    for (int c = 0; c < C; ++c) {
        float a = xi[c], b = xj[c];
        o[c]     = (_Float16)a;
        o[C + c] = (_Float16)(mode ? b : (b - a));
    }
    for (int c = 2 * C; c < Kpad; ++c) o[c] = (_Float16)0.f;
}

// f32 -> f16 with row-padding
__global__ void to_half_kernel(const float* __restrict__ src,
                               _Float16* __restrict__ dst,
                               int rows, int srcStride, int Kcopy, int Kdst)
{
    int t = blockIdx.x * 256 + threadIdx.x;
    if (t >= rows * Kdst) return;
    int r = t / Kdst, c = t - r * Kdst;
    float v = (c < Kcopy) ? src[(size_t)r * srcStride + c] : 0.f;
    dst[t] = (_Float16)v;
}

// ---------------------------------------------------------------------------
// TDM: DMA an Nout x K f16 weight panel (row-major) from global into LDS.
// D# per CDNA5 ISA ch.8: group0 = {count, lds_addr, global_addr, type=2};
// group1 = {data_size=2B, tensor_dim0=K, tensor_dim1=Nout, tile=KxNout,
// dim0_stride=K}. Issued by one wave; completion via TENSORcnt.
// ---------------------------------------------------------------------------
__device__ __forceinline__ void tdm_load_weights(const _Float16* W,
                                                 _Float16* lds, int K, int Nout)
{
    unsigned long long ga = (unsigned long long)(uintptr_t)W;
    unsigned int ldsOff = (unsigned int)(uintptr_t)lds; // flat shared addr low bits = LDS offset
    u32x4 g0;
    g0[0] = 1u;                                            // count=1 (valid user D#)
    g0[1] = ldsOff;                                        // lds_addr (bytes)
    g0[2] = (unsigned int)ga;                              // global_addr[31:0]
    g0[3] = (unsigned int)((ga >> 32) & 0x01FFFFFFull)     // global_addr[56:32]
          | (2u << 30);                                    // type = 2 ("image")
    i32x8 g1;
    g1[0] = 0x10000;                                       // data_size = 2 bytes
    g1[1] = (K & 0xFFFF) << 16;                            // tensor_dim0[15:0]
    g1[2] = (Nout & 0xFFFF) << 16;                         // tensor_dim1[15:0]
    g1[3] = (K & 0xFFFF) << 16;                            // tile_dim0
    g1[4] = (Nout & 0xFFFF);                               // tile_dim1 (tile_dim2=0)
    g1[5] = K;                                             // tensor_dim0_stride[31:0]
    g1[6] = 0;
    g1[7] = 0;
    i32x4 gz = {0, 0, 0, 0};
#if __has_builtin(__builtin_amdgcn_tensor_load_to_lds)
#if __clang_major__ >= 23
    i32x8 gz8 = {0, 0, 0, 0, 0, 0, 0, 0};
    __builtin_amdgcn_tensor_load_to_lds(g0, g1, gz, gz, gz8, 0);
#else
    __builtin_amdgcn_tensor_load_to_lds(g0, g1, gz, gz, 0);
#endif
#else
    asm volatile("tensor_load_to_lds %0, %1, %2, %3"
                 :: "s"(g0), "s"(g1), "s"(gz), "s"(gz) : "memory");
#endif
    __builtin_amdgcn_s_wait_tensorcnt(0);
}

// ---------------------------------------------------------------------------
// Generic WMMA GEMM: out[M,N] = A[M,K] @ W[N,K]^T (+bias / BN+ReLU epilogues).
// One wave per 16x16 tile; K mult of 32, M/N mult of 16, M mult of 128 so no
// wave exits before the staging barrier. stageW: TDM weight panel -> LDS.
// ---------------------------------------------------------------------------
__global__ void wmma_gemm_kernel(const _Float16* __restrict__ A,
                                 const _Float16* __restrict__ W,
                                 const float* __restrict__ bias,
                                 const float* __restrict__ bng,
                                 const float* __restrict__ bnb,
                                 _Float16* __restrict__ outH,
                                 float* __restrict__ outF,
                                 int M, int K, int Nout, int epi, int stageW)
{
    __shared__ _Float16 wlds[kWPanel];              // 48 KB of the 320 KB WGP LDS
    const int lane  = threadIdx.x & 31;
    const int wave  = threadIdx.x >> 5;
    const int tileM = blockIdx.x * 8 + wave;
    const int tileN = blockIdx.y;

    if (stageW) {
        if (wave == 0) tdm_load_weights(W, wlds, K, Nout);
        __syncthreads();                            // release after TENSORcnt==0
    }
    if (tileM * 16 >= M) return;                    // wave-uniform (never before barrier)

    const int hsel = lane >> 4;
    const int lmod = lane & 15;
    const _Float16* Arow = A + (size_t)(tileM * 16 + lmod) * K;
    v8f acc = {};
    if (stageW) {
        const _Float16* Wrow = wlds + (size_t)(tileN * 16 + lmod) * K;
        for (int k0 = 0; k0 < K; k0 += 32) {
            if (k0 + 32 < K) __builtin_prefetch(Arow + k0 + 32, 0, 3);
            union { v16h v; v8h h[2]; } af;
            af.h[0] = *(const v8h*)(Arow + k0 + hsel * 8);
            af.h[1] = *(const v8h*)(Arow + k0 + 16 + hsel * 8);
            union { v16h v; v8h h[2]; } bfu;
            bfu.h[0] = *(const v8h*)(Wrow + k0 + hsel * 8);   // ds_load path
            bfu.h[1] = *(const v8h*)(Wrow + k0 + 16 + hsel * 8);
            v16h bf = bfu.v;
            acc = __builtin_amdgcn_wmma_f32_16x16x32_f16(false, af.v, false, bf,
                                                         (short)0, acc, false, false);
        }
    } else {
        const _Float16* Wrow = W + (size_t)(tileN * 16 + lmod) * K;
        for (int k0 = 0; k0 < K; k0 += 32) {
            if (k0 + 32 < K) __builtin_prefetch(Arow + k0 + 32, 0, 3);
            union { v16h v; v8h h[2]; } af;
            af.h[0] = *(const v8h*)(Arow + k0 + hsel * 8);
            af.h[1] = *(const v8h*)(Arow + k0 + 16 + hsel * 8);
            union { v16h v; v8h h[2]; } bfu;
            bfu.h[0] = *(const v8h*)(Wrow + k0 + hsel * 8);
            bfu.h[1] = *(const v8h*)(Wrow + k0 + 16 + hsel * 8);
            v16h bf = bfu.v;
            acc = __builtin_amdgcn_wmma_f32_16x16x32_f16(false, af.v, false, bf,
                                                         (short)0, acc, false, false);
        }
    }
    const int n  = tileN * 16 + lmod;
    const float bb = bias ? bias[n] : 0.f;
    const float g  = bng ? bng[n] * kBnScale : 1.f;
    const float be = bnb ? bnb[n] : 0.f;
#pragma unroll
    for (int r = 0; r < 8; ++r) {
        int m = tileM * 16 + r + 8 * hsel;
        float v = acc[r] + bb;
        if (epi == 2) v = v * g + be;
        if (epi >= 1) v = fmaxf(v, 0.f);
        size_t o = (size_t)m * Nout + n;
        if (outH) outH[o] = (_Float16)v;
        if (outF) outF[o] = v;
    }
}

// ---------------------------------------------------------------------------
// Edge epilogue: sigmoid-gated modulation, max over KNN, LayerNorm + ReLU.
// ---------------------------------------------------------------------------
__global__ void edge_combine_kernel(const _Float16* __restrict__ h2,
                                    const _Float16* __restrict__ gb,
                                    const float* __restrict__ lng,
                                    const float* __restrict__ lnb,
                                    float* __restrict__ xout,
                                    _Float16* __restrict__ xcat, int catOff)
{
    int p = blockIdx.x * 256 + threadIdx.x;
    if (p >= kNP) return;
    float o[32];
#pragma unroll
    for (int c = 0; c < 32; ++c) o[c] = -1e30f;
    for (int k = 0; k < kK; ++k) {
        size_t r = (size_t)p * kK + k;
        const _Float16* hh = h2 + r * 32;
        const _Float16* gg = gb + r * 64;
#pragma unroll
        for (int c = 0; c < 32; ++c) {
            float s = 1.f / (1.f + expf(-((float)gg[c] + 1.f)));
            float m = s * (float)hh[c] + (float)gg[32 + c];
            o[c] = fmaxf(o[c], m);
        }
    }
    float mean = 0.f;
#pragma unroll
    for (int c = 0; c < 32; ++c) mean += o[c];
    mean *= (1.f / 32.f);
    float var = 0.f;
#pragma unroll
    for (int c = 0; c < 32; ++c) { float d = o[c] - mean; var += d * d; }
    var *= (1.f / 32.f);
    float inv = rsqrtf(var + kEps);
#pragma unroll
    for (int c = 0; c < 32; ++c) {
        float v = fmaxf((o[c] - mean) * inv * lng[c] + lnb[c], 0.f);
        xout[(size_t)p * 32 + c] = v;
        xcat[(size_t)p * 96 + catOff + c] = (_Float16)v;
    }
}

__global__ void frame_max_kernel(const float* __restrict__ xin, float* __restrict__ E)
{
    int t = blockIdx.x * 256 + threadIdx.x;
    if (t >= kBT * 256) return;
    int bt = t >> 8, d = t & 255;
    const float* base = xin + (size_t)bt * kN * 256 + d;
    float m = -1e30f;
    for (int n = 0; n < kN; ++n) m = fmaxf(m, base[(size_t)n * 256]);
    E[t] = m;
}

__global__ void cross_attn_kernel(const float* __restrict__ E, const float* __restrict__ fs,
    const float* g0, const float* b0, const float* g1, const float* b1,
    const float* g2, const float* b2,
    const float* __restrict__ Wq, const float* __restrict__ Wk,
    const float* __restrict__ Wv, const float* __restrict__ Wo,
    const float* __restrict__ pos, float* __restrict__ seq)
{
    __shared__ float sv[9];
    __shared__ float av[64];
    const int bt = blockIdx.x, t = threadIdx.x;
    if (t < 3) {
        const float* c = fs + bt * 9 + 3 * t;
        const float* g = (t == 0) ? g0 : (t == 1) ? g1 : g2;
        const float* b = (t == 0) ? b0 : (t == 1) ? b1 : b2;
        float m = (c[0] + c[1] + c[2]) * (1.f / 3.f);
        float v = 0.f;
        for (int i = 0; i < 3; ++i) { float d = c[i] - m; v += d * d; }
        float inv = rsqrtf(v * (1.f / 3.f) + kEps);
        for (int i = 0; i < 3; ++i) sv[3 * t + i] = (c[i] - m) * inv * g[i] + b[i];
    }
    __syncthreads();
    if (t < 64) {
        float q = 0.f;
        for (int d = 0; d < 256; ++d) q += Wq[t * 256 + d] * E[bt * 256 + d];
        float kk = 0.f, vv = 0.f;
        for (int d = 0; d < 9; ++d) { kk += Wk[t * 9 + d] * sv[d]; vv += Wv[t * 9 + d] * sv[d]; }
        float attn = 1.f / (1.f + expf(-(q * kk * 0.125f)));
        av[t] = attn * vv;
    }
    __syncthreads();
    int d = t;
    float acc = 0.f;
    for (int c = 0; c < 64; ++c) acc += av[c] * Wo[d * 64 + c];
    int tt = bt & 31;
    seq[bt * 256 + d] = E[bt * 256 + d] + acc + pos[tt * 256 + d];
}

__global__ void layernorm_kernel(const float* __restrict__ x, const float* __restrict__ g,
                                 const float* __restrict__ b, float* __restrict__ y)
{
    __shared__ float red[256];
    const int row = blockIdx.x, t = threadIdx.x;
    float v = x[(size_t)row * 256 + t];
    red[t] = v; __syncthreads();
    for (int s = 128; s > 0; s >>= 1) { if (t < s) red[t] += red[t + s]; __syncthreads(); }
    float mean = red[0] * (1.f / 256.f); __syncthreads();
    float dv = v - mean;
    red[t] = dv * dv; __syncthreads();
    for (int s = 128; s > 0; s >>= 1) { if (t < s) red[t] += red[t + s]; __syncthreads(); }
    float inv = rsqrtf(red[0] * (1.f / 256.f) + kEps);
    y[(size_t)row * 256 + t] = dv * inv * g[t] + b[t];
}

__global__ void conv_silu_kernel(const float* __restrict__ xz, const float* __restrict__ cw,
                                 const float* __restrict__ cb, float* __restrict__ xcs)
{
    int t = blockIdx.x * 256 + threadIdx.x;
    if (t >= kBT * 512) return;
    int d = t & 511, row = t >> 9, l = row & 31, b = row >> 5;
    float acc = cb[d];
#pragma unroll
    for (int i = 0; i < 4; ++i) {
        int ls = l - 3 + i;
        if (ls >= 0) acc += xz[((size_t)(b * 32 + ls)) * 1024 + d] * cw[d * 4 + i];
    }
    xcs[t] = acc / (1.f + expf(-acc));
}

__global__ void softplus_kernel(float* __restrict__ x, int n)
{
    int t = blockIdx.x * 256 + threadIdx.x;
    if (t >= n) return;
    float v = x[t];
    x[t] = (v > 20.f) ? v : log1pf(expf(v));
}

__global__ void ssm_scan_kernel(const float* __restrict__ dt, const float* __restrict__ dbl,
                                const float* __restrict__ xcs, const float* __restrict__ xz,
                                const float* __restrict__ A_log, const float* __restrict__ Dp,
                                float* __restrict__ y)
{
    int t = blockIdx.x * 256 + threadIdx.x;
    if (t >= kB * 512) return;
    int d = t & 511, b = t >> 9;
    float A[16], h[16];
#pragma unroll
    for (int n = 0; n < 16; ++n) { A[n] = -expf(A_log[d * 16 + n]); h[n] = 0.f; }
    float Dd = Dp[d];
    for (int l = 0; l < 32; ++l) {
        int row = b * 32 + l;
        float dtv = dt[(size_t)row * 512 + d];
        float xv  = xcs[(size_t)row * 512 + d];
        float acc = 0.f;
#pragma unroll
        for (int n = 0; n < 16; ++n) {
            float Bv = dbl[(size_t)row * 48 + 16 + n];
            float Cv = dbl[(size_t)row * 48 + 32 + n];
            h[n] = expf(dtv * A[n]) * h[n] + dtv * xv * Bv;
            acc += h[n] * Cv;
        }
        float yv = acc + Dd * xv;
        float z = xz[(size_t)row * 1024 + 512 + d];
        yv *= z / (1.f + expf(-z));
        y[(size_t)row * 512 + d] = yv;
    }
}

__global__ void add_kernel(const float* __restrict__ a, const float* __restrict__ b,
                           float* __restrict__ c, int n)
{
    int t = blockIdx.x * 256 + threadIdx.x;
    if (t < n) c[t] = a[t] + b[t];
}

__global__ void mean_kernel(const float* __restrict__ seq, float* __restrict__ feat)
{
    int t = blockIdx.x * 256 + threadIdx.x;
    if (t >= kB * 256) return;
    int b = t >> 8, d = t & 255;
    float s = 0.f;
    for (int l = 0; l < 32; ++l) s += seq[(size_t)(b * 32 + l) * 256 + d];
    feat[t] = s * (1.f / 32.f);
}

__global__ void head_kernel(const float* __restrict__ feat,
    const float* w0, const float* b0, const float* w1, const float* b1,
    const float* w2, const float* b2, const float* w3, const float* b3,
    float* __restrict__ out)
{
    __shared__ float h0[256], h1[256];
    const int t = threadIdx.x;
    for (int bb = 0; bb < kB; ++bb) {
        const float* f = feat + bb * 256;
        { float acc = b0[t]; for (int k = 0; k < 256; ++k) acc += w0[t * 256 + k] * f[k];
          h0[t] = fmaxf(acc, 0.f); }
        __syncthreads();
        if (t < 128) { float acc = b1[t]; for (int k = 0; k < 256; ++k) acc += w1[t * 256 + k] * h0[k];
                       h1[t] = fmaxf(acc, 0.f); }
        __syncthreads();
        if (t < 64) { float acc = b2[t]; for (int k = 0; k < 128; ++k) acc += w2[t * 128 + k] * h1[k];
                      h0[t] = fmaxf(acc, 0.f); }
        __syncthreads();
        if (t < 9) { float acc = b3[t]; for (int k = 0; k < 64; ++k) acc += w3[t * 64 + k] * h0[k];
                     out[bb * 9 + t] = acc; }
        __syncthreads();
    }
}

// ---------------------------------------------------------------------------
// Host orchestration
// ---------------------------------------------------------------------------
extern "C" void kernel_launch(void* const* d_in, const int* in_sizes, int n_in,
                              void* d_out, int out_size, void* d_ws, size_t ws_size,
                              hipStream_t stream)
{
    (void)in_sizes; (void)n_in; (void)out_size; (void)ws_size;
    auto F = [&](int i) { return (const float*)d_in[i]; };

    const float* point_cloud   = F(0);
    const float* frame_signals = F(1);
    int pi = 2;
    struct EdgeP { const float *m0w,*m0b,*m0g,*m0be,*m1w,*m1b,*m1g,*m1be,
                               *aw1,*ab1,*aw2,*ab2,*lng,*lnb; };
    EdgeP ep[3];
    for (int l = 0; l < 3; ++l) {
        ep[l].m0w = F(pi++); ep[l].m0b = F(pi++); ep[l].m0g = F(pi++); ep[l].m0be = F(pi++);
        ep[l].m1w = F(pi++); ep[l].m1b = F(pi++); ep[l].m1g = F(pi++); ep[l].m1be = F(pi++);
        ep[l].aw1 = F(pi++); ep[l].ab1 = F(pi++); ep[l].aw2 = F(pi++); ep[l].ab2 = F(pi++);
        ep[l].lng = F(pi++); ep[l].lnb = F(pi++);
    }
    const float* lin1_w  = F(pi++); const float* lin1_b  = F(pi++);
    const float* lin1_g  = F(pi++); const float* lin1_be = F(pi++);
    const float *lng_[3], *lnb_[3];
    for (int i = 0; i < 3; ++i) { lng_[i] = F(pi++); lnb_[i] = F(pi++); }
    const float* Wq = F(pi++); const float* Wk = F(pi++);
    const float* Wv = F(pi++); const float* Wo = F(pi++);
    const float* pos = F(pi++);
    struct SsmP { const float *lng,*lnb,*in_w,*conv_w,*conv_b,*xproj_w,*dt_w,*dt_b,
                              *A_log,*Dp,*out_w; };
    SsmP sp[2];
    for (int i = 0; i < 2; ++i) {
        sp[i].lng = F(pi++); sp[i].lnb = F(pi++); sp[i].in_w = F(pi++);
        sp[i].conv_w = F(pi++); sp[i].conv_b = F(pi++); sp[i].xproj_w = F(pi++);
        sp[i].dt_w = F(pi++); sp[i].dt_b = F(pi++); sp[i].A_log = F(pi++);
        sp[i].Dp = F(pi++); sp[i].out_w = F(pi++);
    }
    const float *ow0 = F(pi++), *ob0 = F(pi++), *ow1 = F(pi++), *ob1 = F(pi++);
    const float *ow2 = F(pi++), *ob2 = F(pi++), *ow3 = F(pi++), *ob3 = F(pi++);

    char* wsp = (char*)d_ws;
    size_t off = 0;
    auto alloc = [&](size_t bytes) -> void* {
        off = (off + 255) & ~(size_t)255;
        void* p = wsp + off; off += bytes; return p;
    };
    int*      idxb  = (int*)     alloc((size_t)kMEdge * 4);
    float*    xAbuf = (float*)   alloc((size_t)kNP * 32 * 4);
    float*    xBbuf = (float*)   alloc((size_t)kNP * 32 * 4);
    _Float16* xcat  = (_Float16*)alloc((size_t)kNP * 96 * 2);
    _Float16* regA  = (_Float16*)alloc((size_t)kMEdge * 64 * 2); // ef, then aux-hidden
    _Float16* regBC = (_Float16*)alloc((size_t)kMEdge * 64 * 2); // ea+h1, then gb
    _Float16* regD  = (_Float16*)alloc((size_t)kMEdge * 32 * 2); // h2
    float*    lin1o = (float*)   alloc((size_t)kNP * 256 * 4);
    float*    Ebuf  = (float*)   alloc((size_t)kBT * 256 * 4);
    float*    seq   = (float*)   alloc((size_t)kBT * 256 * 4);
    float*    xln   = (float*)   alloc((size_t)kBT * 256 * 4);
    float*    xz    = (float*)   alloc((size_t)kBT * 1024 * 4);
    float*    xcs   = (float*)   alloc((size_t)kBT * 512 * 4);
    float*    dblb  = (float*)   alloc((size_t)kBT * 48 * 4);
    float*    dtb   = (float*)   alloc((size_t)kBT * 512 * 4);
    float*    ybuf  = (float*)   alloc((size_t)kBT * 512 * 4);
    float*    moutb = (float*)   alloc((size_t)kBT * 256 * 4);
    float*    feat  = (float*)   alloc((size_t)kB * 256 * 4);
    _Float16* actH  = (_Float16*)alloc((size_t)kBT * 1024 * 2);
    _Float16* wh    = (_Float16*)alloc((size_t)1024 * 512 * 2);
    _Float16* h1    = regBC + (size_t)kMEdge * 32;

    const dim3 blk(256);
    auto g1d = [](int n) { return dim3((n + 255) / 256); };

    // ---------------- Edge-conv stack (3 layers) ----------------
    for (int l = 0; l < 3; ++l) {
        const float* xs; int xstride, xoffs, C;
        if (l == 0) { xs = point_cloud; xstride = 6; xoffs = 0; C = 3; }
        else        { xs = (l == 1) ? xAbuf : xBbuf; xstride = 32; xoffs = 0; C = 32; }
        float* xo = (l == 0) ? xAbuf : ((l == 1) ? xBbuf : xAbuf);
        const int Kef = (l == 0) ? 32 : 64;

        knn_kernel<<<dim3(kN / 8, kBT), blk, 0, stream>>>(xs, xstride, xoffs, C, idxb);
        build_feat_kernel<<<g1d(kMEdge), blk, 0, stream>>>(xs, xstride, xoffs, C, idxb, regA, Kef, 0);
        build_feat_kernel<<<g1d(kMEdge), blk, 0, stream>>>(point_cloud, 6, 3, 3, idxb, regBC, 32, 1);

        to_half_kernel<<<g1d(32 * Kef), blk, 0, stream>>>(ep[l].m0w, wh, 32, 2 * C, 2 * C, Kef);
        wmma_gemm_kernel<<<dim3(kMEdge / 128, 2), blk, 0, stream>>>(
            regA, wh, ep[l].m0b, ep[l].m0g, ep[l].m0be, h1, nullptr, kMEdge, Kef, 32, 2, 1);
        to_half_kernel<<<g1d(32 * 32), blk, 0, stream>>>(ep[l].m1w, wh, 32, 32, 32, 32);
        wmma_gemm_kernel<<<dim3(kMEdge / 128, 2), blk, 0, stream>>>(
            h1, wh, ep[l].m1b, ep[l].m1g, ep[l].m1be, regD, nullptr, kMEdge, 32, 32, 2, 1);
        to_half_kernel<<<g1d(64 * 32), blk, 0, stream>>>(ep[l].aw1, wh, 64, 6, 6, 32);
        wmma_gemm_kernel<<<dim3(kMEdge / 128, 4), blk, 0, stream>>>(
            regBC, wh, ep[l].ab1, nullptr, nullptr, regA, nullptr, kMEdge, 32, 64, 1, 1);
        to_half_kernel<<<g1d(64 * 64), blk, 0, stream>>>(ep[l].aw2, wh, 64, 64, 64, 64);
        wmma_gemm_kernel<<<dim3(kMEdge / 128, 4), blk, 0, stream>>>(
            regA, wh, ep[l].ab2, nullptr, nullptr, regBC, nullptr, kMEdge, 64, 64, 0, 1);
        edge_combine_kernel<<<g1d(kNP), blk, 0, stream>>>(
            regD, regBC, ep[l].lng, ep[l].lnb, xo, xcat, 32 * l);
    }

    // ---------------- lin1 (96 -> 256) + frame max ----------------
    to_half_kernel<<<g1d(256 * 96), blk, 0, stream>>>(lin1_w, wh, 256, 96, 96, 96);
    wmma_gemm_kernel<<<dim3(kNP / 128, 16), blk, 0, stream>>>(
        xcat, wh, lin1_b, lin1_g, lin1_be, nullptr, lin1o, kNP, 96, 256, 2, 1); // 48KB panel
    frame_max_kernel<<<g1d(kBT * 256), blk, 0, stream>>>(lin1o, Ebuf);

    // ---------------- cross-attention gate + pos ----------------
    cross_attn_kernel<<<dim3(kBT), blk, 0, stream>>>(
        Ebuf, frame_signals, lng_[0], lnb_[0], lng_[1], lnb_[1], lng_[2], lnb_[2],
        Wq, Wk, Wv, Wo, pos, seq);

    // ---------------- 2x Mamba blocks ----------------
    for (int i = 0; i < 2; ++i) {
        const SsmP& s = sp[i];
        layernorm_kernel<<<dim3(kBT), blk, 0, stream>>>(seq, s.lng, s.lnb, xln);
        to_half_kernel<<<g1d(kBT * 256), blk, 0, stream>>>(xln, actH, kBT, 256, 256, 256);
        to_half_kernel<<<g1d(1024 * 256), blk, 0, stream>>>(s.in_w, wh, 1024, 256, 256, 256);
        wmma_gemm_kernel<<<dim3(1, 64), blk, 0, stream>>>(               // 512KB panel: stream
            actH, wh, nullptr, nullptr, nullptr, nullptr, xz, kBT, 256, 1024, 0, 0);
        conv_silu_kernel<<<g1d(kBT * 512), blk, 0, stream>>>(xz, s.conv_w, s.conv_b, xcs);
        to_half_kernel<<<g1d(kBT * 512), blk, 0, stream>>>(xcs, actH, kBT, 512, 512, 512);
        to_half_kernel<<<g1d(48 * 512), blk, 0, stream>>>(s.xproj_w, wh, 48, 512, 512, 512);
        wmma_gemm_kernel<<<dim3(1, 3), blk, 0, stream>>>(                // 48KB panel
            actH, wh, nullptr, nullptr, nullptr, nullptr, dblb, kBT, 512, 48, 0, 1);
        to_half_kernel<<<g1d(kBT * 32), blk, 0, stream>>>(dblb, actH, kBT, 48, 16, 32);
        to_half_kernel<<<g1d(512 * 32), blk, 0, stream>>>(s.dt_w, wh, 512, 16, 16, 32);
        wmma_gemm_kernel<<<dim3(1, 32), blk, 0, stream>>>(               // 32KB panel
            actH, wh, s.dt_b, nullptr, nullptr, nullptr, dtb, kBT, 32, 512, 0, 1);
        softplus_kernel<<<g1d(kBT * 512), blk, 0, stream>>>(dtb, kBT * 512);
        ssm_scan_kernel<<<g1d(kB * 512), blk, 0, stream>>>(
            dtb, dblb, xcs, xz, s.A_log, s.Dp, ybuf);
        to_half_kernel<<<g1d(kBT * 512), blk, 0, stream>>>(ybuf, actH, kBT, 512, 512, 512);
        to_half_kernel<<<g1d(256 * 512), blk, 0, stream>>>(s.out_w, wh, 256, 512, 512, 512);
        wmma_gemm_kernel<<<dim3(1, 16), blk, 0, stream>>>(               // 256KB panel: stream
            actH, wh, nullptr, nullptr, nullptr, nullptr, moutb, kBT, 512, 256, 0, 0);
        add_kernel<<<g1d(kBT * 256), blk, 0, stream>>>(seq, moutb, seq, kBT * 256);
    }

    // ---------------- mean over T + head MLP ----------------
    mean_kernel<<<g1d(kB * 256), blk, 0, stream>>>(seq, feat);
    head_kernel<<<dim3(1), blk, 0, stream>>>(feat, ow0, ob0, ow1, ob1, ow2, ob2, ow3, ob3,
                                             (float*)d_out);
}
